// PointNetEncoder_11433202942871
// MI455X (gfx1250) — matrix operand order repaired and checked by reading, not compiled, not run on hardware
//
#include <hip/hip_runtime.h>
#include <hip/hip_bf16.h>

// ---------------------------------------------------------------------------
// PointNet++ segmentation forward on gfx1250 (MI455X).
// All 1x1-conv chains map to f16 WMMA GEMMs (v_wmma_f32_16x16x32_f16).
// f16 activations are stored COLUMN-major (X[col*Kp + k]) so both A and B
// WMMA fragments load as b128 vectors; f32 pre-BN outputs stay ROW-major so
// the batchnorm reductions are coalesced. BN bias cancellation lets us skip
// conv biases everywhere except the final head layer.
// ---------------------------------------------------------------------------

typedef _Float16 half_t;
typedef _Float16 h16v __attribute__((ext_vector_type(16)));
typedef _Float16 h8v  __attribute__((ext_vector_type(8)));
typedef float    f8v  __attribute__((ext_vector_type(8)));

#define BB   16
#define NN   16384
#define KCLS 128

// ------------------------------ small utils --------------------------------

__global__ void k_pack_w(const float* __restrict__ W, half_t* __restrict__ Wp,
                         int M, int K, int Mp, int Kp) {
    int i = blockIdx.x * 256 + threadIdx.x;
    if (i >= Mp * Kp) return;
    int m = i / Kp, k = i - m * Kp;
    float v = (m < M && k < K) ? W[m * K + k] : 0.f;
    Wp[i] = (half_t)v;
}

// x (B,5,N) -> ptsT (B*N, 5)
__global__ void k_transpose_pts(const float* __restrict__ x, float* __restrict__ ptsT) {
    int i = blockIdx.x * 256 + threadIdx.x;  // b*N + n
    if (i >= BB * NN) return;
    int b = i >> 14, n = i & (NN - 1);
    const float* src = x + (size_t)b * 5 * NN + n;
#pragma unroll
    for (int c = 0; c < 5; ++c) ptsT[(size_t)i * 5 + c] = src[(size_t)c * NN];
}

// ----------------------- farthest point sampling ---------------------------
__global__ void k_fps(const float* __restrict__ ptsT, int S, int npoint,
                      int* __restrict__ fidx, float* __restrict__ distws) {
    int b = blockIdx.x;
    int tid = threadIdx.x;
    const int T = 256;
    const float* pts = ptsT + (size_t)b * S * 5;
    float* dist = distws + (size_t)b * S;
    for (int i = tid; i < S; i += T) dist[i] = 1e10f;
    __shared__ float sval[256];
    __shared__ int   sidx[256];
    __shared__ int   sfar;
    if (tid == 0) sfar = 0;
    __syncthreads();
    for (int it = 0; it < npoint; ++it) {
        int far = sfar;
        if (tid == 0) fidx[b * npoint + it] = far;
        float c0 = pts[far * 5 + 0], c1 = pts[far * 5 + 1], c2 = pts[far * 5 + 2];
        float c3 = pts[far * 5 + 3], c4 = pts[far * 5 + 4];
        float bv = -1.f; int bi = 0;
        for (int i = tid; i < S; i += T) {
            const float* p = pts + (size_t)i * 5;
            float d0 = p[0] - c0, d1 = p[1] - c1, d2 = p[2] - c2, d3 = p[3] - c3, d4 = p[4] - c4;
            float d = d0 * d0 + d1 * d1 + d2 * d2 + d3 * d3 + d4 * d4;
            float nd = fminf(dist[i], d);
            dist[i] = nd;
            if (nd > bv || (nd == bv && i < bi)) { bv = nd; bi = i; }
        }
        sval[tid] = bv; sidx[tid] = bi;
        __syncthreads();
        for (int o = 128; o > 0; o >>= 1) {
            if (tid < o) {
                float v2 = sval[tid + o]; int i2 = sidx[tid + o];
                if (v2 > sval[tid] || (v2 == sval[tid] && i2 < sidx[tid])) { sval[tid] = v2; sidx[tid] = i2; }
            }
            __syncthreads();
        }
        if (tid == 0) sfar = sidx[0];
        __syncthreads();
    }
}

__global__ void k_gather_xyz(float* __restrict__ dst, const float* __restrict__ srcT,
                             const int* __restrict__ fidx, int S, int Np) {
    int t = blockIdx.x * 256 + threadIdx.x;
    if (t >= BB * Np * 5) return;
    int pt = t / 5, c = t - pt * 5;
    int b = pt / Np;
    dst[t] = srcT[((size_t)b * S + fidx[pt]) * 5 + c];
}

// Ball query: first 8 indices (ascending) with d<=r2, padded with first.
__global__ void k_ball(const float* __restrict__ srcT, const float* __restrict__ newT,
                       int S, int Np, float r2, int* __restrict__ gidx) {
    int t = blockIdx.x * 256 + threadIdx.x;
    if (t >= BB * Np) return;
    int b = t / Np;
    const float* q = newT + (size_t)t * 5;
    float q0 = q[0], q1 = q[1], q2 = q[2], q3 = q[3], q4 = q[4];
    const float* base = srcT + (size_t)b * S * 5;
    int out[8]; int cnt = 0;
    for (int i = 0; i < S && cnt < 8; ++i) {
        const float* p = base + (size_t)i * 5;
        float a0 = p[0] - q0, a1 = p[1] - q1, a2 = p[2] - q2, a3 = p[3] - q3, a4 = p[4] - q4;
        float d = a0 * a0 + a1 * a1 + a2 * a2 + a3 * a3 + a4 * a4;
        if (d <= r2) out[cnt++] = i;
    }
    int f = (cnt > 0) ? out[0] : 0;
    for (int j = cnt; j < 8; ++j) out[j] = f;
    for (int j = 0; j < 8; ++j) gidx[t * 8 + j] = out[j];
}

// --------------------- grouping (column-major f16 out) ---------------------
// SA1: stride 32 (K=10 padded), cols 4096 = (b,p,s); Xo[col*32 + c]
__global__ void k_group1(const float* __restrict__ ptsT, const float* __restrict__ l1x,
                         const int* __restrict__ gidx, half_t* __restrict__ Xo) {
    int i = blockIdx.x * 256 + threadIdx.x;
    if (i >= 4096 * 32) return;
    int col = i >> 5, c = i & 31;
    int b = col >> 8, rem = col & 255, p = rem >> 3, s = rem & 7;
    float v = 0.f;
    if (c < 10) {
        int g = gidx[((b * 32 + p) << 3) + s];
        if (c < 5) v = ptsT[((size_t)b * NN + g) * 5 + c] - l1x[((size_t)b * 32 + p) * 5 + c];
        else       v = ptsT[((size_t)b * NN + g) * 5 + (c - 5)];
    }
    Xo[i] = (half_t)v;
}

// SA2: stride 160 (K=133 padded), cols 2048; P1 is (point, 128) contiguous
__global__ void k_group2(const float* __restrict__ l1x, const float* __restrict__ l2x,
                         const half_t* __restrict__ P1, const int* __restrict__ gidx,
                         half_t* __restrict__ Xo) {
    int i = blockIdx.x * 256 + threadIdx.x;
    if (i >= 2048 * 160) return;
    int col = i / 160, c = i - col * 160;
    int b = col >> 7, rem = col & 127, p = rem >> 3, s = rem & 7;
    float v = 0.f;
    if (c < 133) {
        int g = gidx[((b * 16 + p) << 3) + s];
        if (c < 5) v = l1x[((size_t)b * 32 + g) * 5 + c] - l2x[((size_t)b * 16 + p) * 5 + c];
        else       v = (float)P1[(size_t)(b * 32 + g) * 128 + (c - 5)];
    }
    Xo[i] = (half_t)v;
}

// SA3 (group_all): stride 288 (K=261 padded), cols 256 = (b, s)
__global__ void k_group3(const float* __restrict__ l2x, const half_t* __restrict__ P2,
                         half_t* __restrict__ Xo) {
    int i = blockIdx.x * 256 + threadIdx.x;
    if (i >= 256 * 288) return;
    int col = i / 288, c = i - col * 288;
    float v = 0.f;
    if (c < 5)        v = l2x[(size_t)col * 5 + c];
    else if (c < 261) v = (float)P2[(size_t)col * 256 + (c - 5)];
    Xo[i] = (half_t)v;
}

// ------------------------------ WMMA GEMM ----------------------------------
// Y(Mp16, Cols) row-major = Wp(Mp16, Kp) row-major * Xt(Cols, Kp) col-major.
// One wave computes a 16x16 tile. A-frag per lane: row = lane%16,
// k = k0 + hi*16 + 0..15 contiguous (two b128 loads). B-frag per lane:
// col = lane%16, k = k0 + hi*16 + 0..15 contiguous (two b128 loads).
// D: row = r + 8*hi, col = lane%16.
__global__ __launch_bounds__(32) void k_gemm(const half_t* __restrict__ Wp,
                                             const half_t* __restrict__ Xt,
                                             float* __restrict__ Y,
                                             int Kp, int Cols) {
    int tn = blockIdx.x, tm = blockIdx.y;
    int lane = threadIdx.x;
    int l16 = lane & 15, hi = lane >> 4;
    f8v acc = {0.f, 0.f, 0.f, 0.f, 0.f, 0.f, 0.f, 0.f};
    const half_t* arow = Wp + ((size_t)(tm * 16 + l16)) * Kp + hi * 16;
    const half_t* bcol = Xt + ((size_t)(tn * 16 + l16)) * Kp + hi * 16;
    for (int k0 = 0; k0 < Kp; k0 += 32) {
        h8v alo = *(const h8v*)(arow);
        h8v ahi = *(const h8v*)(arow + 8);
        h8v blo = *(const h8v*)(bcol);
        h8v bhi = *(const h8v*)(bcol + 8);
        h16v a, bf;
#pragma unroll
        for (int j = 0; j < 8; ++j) {
            a[j] = alo[j]; a[8 + j] = ahi[j];
            bf[j] = blo[j]; bf[8 + j] = bhi[j];
        }
        acc = __builtin_amdgcn_wmma_f32_16x16x32_f16(false, a, false, bf,
                                                     (short)0, acc, false, false);
        arow += 32;
        bcol += 32;
    }
    float* yo = Y + ((size_t)(tm * 16 + hi * 8)) * Cols + tn * 16 + l16;
#pragma unroll
    for (int r = 0; r < 8; ++r) yo[(size_t)r * Cols] = acc[r];
}

// ------------------------- batchnorm + relu --------------------------------
__global__ void k_bnstats(const float* __restrict__ Y, float* __restrict__ mean,
                          float* __restrict__ rstd, int Cols) {
    int m = blockIdx.x;
    const float* row = Y + (size_t)m * Cols;
    __shared__ float ss[256], sq[256];
    float s = 0.f, q = 0.f;
    for (int i = threadIdx.x; i < Cols; i += 256) { float v = row[i]; s += v; q += v * v; }
    ss[threadIdx.x] = s; sq[threadIdx.x] = q;
    __syncthreads();
    for (int o = 128; o > 0; o >>= 1) {
        if (threadIdx.x < o) { ss[threadIdx.x] += ss[threadIdx.x + o]; sq[threadIdx.x] += sq[threadIdx.x + o]; }
        __syncthreads();
    }
    if (threadIdx.x == 0) {
        float mu = ss[0] / (float)Cols;
        float var = sq[0] / (float)Cols - mu * mu;
        mean[m] = mu;
        rstd[m] = rsqrtf(var + 1e-5f);
    }
}

// BN + ReLU + f16 convert + transpose to column-major via 32x32 LDS tile.
// Grid: (Cols/32, rowsP/32); Cols and rowsP are always multiples of 32.
__global__ void k_bnrelu_t(const float* __restrict__ Y, const float* __restrict__ g,
                           const float* __restrict__ bt, const float* __restrict__ mean,
                           const float* __restrict__ rstd, half_t* __restrict__ Xo,
                           int M, int rowsP, int Cols) {
    __shared__ half_t tile[32][33];
    int colBase = blockIdx.x * 32;
    int rowBase = blockIdx.y * 32;
    int tc = threadIdx.x & 31;
    int tr = threadIdx.x >> 5;  // 0..7
#pragma unroll
    for (int it = 0; it < 4; ++it) {
        int lr = tr + it * 8;
        int r = rowBase + lr;
        float o = 0.f;
        if (r < M) {
            float v = Y[(size_t)r * Cols + colBase + tc];
            v = g[r] * (v - mean[r]) * rstd[r] + bt[r];
            o = v > 0.f ? v : 0.f;
        }
        tile[lr][tc] = (half_t)o;
    }
    __syncthreads();
#pragma unroll
    for (int it = 0; it < 4; ++it) {
        int lc = tr + it * 8;
        Xo[(size_t)(colBase + lc) * rowsP + rowBase + tc] = tile[tc][lc];
    }
}

// max over contiguous groups of gsz columns; X, P column-major
__global__ void k_maxpool(const half_t* __restrict__ X, half_t* __restrict__ P,
                          int rowsP, int groups, int gsz) {
    int i = blockIdx.x * 256 + threadIdx.x;
    if (i >= rowsP * groups) return;
    int g = i / rowsP, m = i - g * rowsP;
    const half_t* src = X + (size_t)g * gsz * rowsP + m;
    float mx = -3.4e38f;
    for (int s = 0; s < gsz; ++s) mx = fmaxf(mx, (float)src[(size_t)s * rowsP]);
    P[i] = (half_t)mx;
}

// strided row-block copy between column-major activations
__global__ void k_copy_rows(const half_t* __restrict__ src, half_t* __restrict__ dst,
                            int rows, int srcStride, int dstStride, int rowOff, int Cols) {
    int i = blockIdx.x * 256 + threadIdx.x;
    if (i >= rows * Cols) return;
    int col = i / rows, c = i - col * rows;
    dst[(size_t)col * dstStride + rowOff + c] = src[(size_t)col * srcStride + c];
}

// fp3 broadcast: l3 features (per batch) into Xfp3 rows 256..767
__global__ void k_bcast_f3(const half_t* __restrict__ F3, half_t* __restrict__ Xfp3) {
    int i = blockIdx.x * 256 + threadIdx.x;
    if (i >= 256 * 512) return;
    int col = i >> 9, m = i & 511;
    int b = col >> 4;
    Xfp3[(size_t)col * 768 + 256 + m] = F3[(size_t)b * 512 + m];
}

// 3-NN with inverse-distance weights (matches lax.top_k tie-breaking)
__global__ void k_knn3(const float* __restrict__ qT, const float* __restrict__ rT,
                       int Nq, int S, int* __restrict__ idx3, float* __restrict__ w3) {
    __shared__ float sp[32 * 5];
    int b = blockIdx.y;
    int n = blockIdx.x * 256 + threadIdx.x;
    if (threadIdx.x < S * 5) sp[threadIdx.x] = rT[(size_t)b * S * 5 + threadIdx.x];
    __syncthreads();
    if (n >= Nq) return;
    int t = b * Nq + n;
    const float* q = qT + (size_t)t * 5;
    float q0 = q[0], q1 = q[1], q2 = q[2], q3 = q[3], q4 = q[4];
    float d0 = 1e30f, d1 = 1e30f, d2 = 1e30f;
    int i0 = 0, i1 = 0, i2 = 0;
    for (int i = 0; i < S; ++i) {
        const float* p = sp + i * 5;
        float a0 = p[0] - q0, a1 = p[1] - q1, a2 = p[2] - q2, a3 = p[3] - q3, a4 = p[4] - q4;
        float d = a0 * a0 + a1 * a1 + a2 * a2 + a3 * a3 + a4 * a4;
        if (d < d0)      { d2 = d1; i2 = i1; d1 = d0; i1 = i0; d0 = d; i0 = i; }
        else if (d < d1) { d2 = d1; i2 = i1; d1 = d;  i1 = i; }
        else if (d < d2) { d2 = d;  i2 = i; }
    }
    float w0 = 1.f / (d0 + 1e-8f), w1 = 1.f / (d1 + 1e-8f), w2 = 1.f / (d2 + 1e-8f);
    float s = w0 + w1 + w2;
    idx3[t * 3 + 0] = i0; idx3[t * 3 + 1] = i1; idx3[t * 3 + 2] = i2;
    w3[t * 3 + 0] = w0 / s; w3[t * 3 + 1] = w1 / s; w3[t * 3 + 2] = w2 / s;
}

// interp: Xo[col*dstStride + rowOff + m] = sum_j w_j * F[(b*Sf+idx_j)*Fs + m]
__global__ void k_interp_build(const half_t* __restrict__ F, int Fs,
                               const int* __restrict__ idx3, const float* __restrict__ w3,
                               half_t* __restrict__ Xo, int dstStride, int rowOff,
                               int rows, int Nq, int Sf) {
    long total = (long)rows * BB * Nq;
    long i = (long)blockIdx.x * 256 + threadIdx.x;
    if (i >= total) return;
    int col = (int)(i / rows);
    int m = (int)(i - (long)col * rows);
    int b = col / Nq;
    long t = col;
    float v = w3[t * 3 + 0] * (float)F[(size_t)(b * Sf + idx3[t * 3 + 0]) * Fs + m]
            + w3[t * 3 + 1] * (float)F[(size_t)(b * Sf + idx3[t * 3 + 1]) * Fs + m]
            + w3[t * 3 + 2] * (float)F[(size_t)(b * Sf + idx3[t * 3 + 2]) * Fs + m];
    Xo[(size_t)col * dstStride + rowOff + m] = (half_t)v;
}

// final: logits (128, B*N) row-major + bias -> log_softmax -> out (B, N, 128)
__global__ void k_logsoftmax(const float* __restrict__ Y, const float* __restrict__ bias,
                             float* __restrict__ out) {
    int col = blockIdx.x * 256 + threadIdx.x;
    if (col >= BB * NN) return;
    const float* yc = Y + col;
    float mx = -3.4e38f;
    for (int k = 0; k < KCLS; ++k) mx = fmaxf(mx, yc[(size_t)k * BB * NN] + bias[k]);
    float s = 0.f;
    for (int k = 0; k < KCLS; ++k) s += expf(yc[(size_t)k * BB * NN] + bias[k] - mx);
    float ls = logf(s) + mx;
    float* o = out + (size_t)col * KCLS;
    for (int k = 0; k < KCLS; ++k) o[k] = yc[(size_t)k * BB * NN] + bias[k] - ls;
}

// ------------------------------- host side ---------------------------------

static inline int ru(int v, int m) { return (v + m - 1) / m * m; }

struct Bump {
    char* base; size_t off;
    void* get(size_t bytes) {
        size_t o = (off + 255) & ~(size_t)255;
        off = o + bytes;
        return base + o;
    }
};

extern "C" void kernel_launch(void* const* d_in, const int* in_sizes, int n_in,
                              void* d_out, int out_size, void* d_ws, size_t ws_size,
                              hipStream_t stream) {
    (void)in_sizes; (void)n_in; (void)out_size; (void)ws_size;
    const float* x = (const float*)d_in[0];
    auto IN = [&](int i) { return (const float*)d_in[i]; };

    Bump ws{(char*)d_ws, 0};

    // --- big buffers (~282MB) ---
    float*  Ybig   = (float*)ws.get((size_t)128 * BB * NN * 4);
    half_t* Xbig_a = (half_t*)ws.get((size_t)128 * BB * NN * 2);
    half_t* Xbig_b = (half_t*)ws.get((size_t)128 * BB * NN * 2);
    float*  ptsT0  = (float*)ws.get((size_t)BB * NN * 5 * 4);
    float*  distws = (float*)ws.get((size_t)BB * NN * 4);
    int*    idx3b  = (int*)ws.get((size_t)BB * NN * 3 * 4);
    float*  w3b    = (float*)ws.get((size_t)BB * NN * 3 * 4);

    // --- small buffers (column-major activations: count = stride * cols) ---
    int*    fidx1 = (int*)ws.get(BB * 32 * 4);
    int*    fidx2 = (int*)ws.get(BB * 16 * 4);
    float*  l1x   = (float*)ws.get(BB * 32 * 5 * 4);
    float*  l2x   = (float*)ws.get(BB * 16 * 5 * 4);
    int*    gidx1 = (int*)ws.get(BB * 32 * 8 * 4);
    int*    gidx2 = (int*)ws.get(BB * 16 * 8 * 4);
    half_t* Xsa1  = (half_t*)ws.get((size_t)32 * 4096 * 2);
    half_t* Xsa2  = (half_t*)ws.get((size_t)160 * 2048 * 2);
    half_t* Xsa3  = (half_t*)ws.get((size_t)288 * 256 * 2);
    half_t* Xfp3  = (half_t*)ws.get((size_t)768 * 256 * 2);
    half_t* Xfp2  = (half_t*)ws.get((size_t)384 * 512 * 2);
    half_t* Xs_a  = (half_t*)ws.get(1 << 20);   // up to 524288 halves
    half_t* Xs_b  = (half_t*)ws.get(1 << 20);
    float*  Ysml  = (float*)ws.get(1 << 21);    // up to 524288 floats
    half_t* P1    = (half_t*)ws.get((size_t)512 * 128 * 2);  // (point, 128)
    half_t* P2    = (half_t*)ws.get((size_t)256 * 256 * 2);  // (point, 256)
    half_t* F3    = (half_t*)ws.get((size_t)16 * 512 * 2);   // (batch, 512)
    half_t* L2F   = (half_t*)ws.get((size_t)256 * 256 * 2);  // (point, 256)
    half_t* L1F   = (half_t*)ws.get((size_t)512 * 128 * 2);  // (point, 128)
    int*    idx3s = (int*)ws.get(BB * 32 * 3 * 4);
    float*  w3s   = (float*)ws.get(BB * 32 * 3 * 4);
    float*  meanb = (float*)ws.get(1024 * 4);
    float*  rstdb = (float*)ws.get(1024 * 4);

    // --- weight table (input index, M=cout, K=cin), in use order ---
    struct WDesc { int idx, M, K; };
    const WDesc wd[20] = {
        {50, 64, 10},  {52, 64, 64},   {54, 128, 64},    // sa1
        {62, 128, 133},{64, 128, 128}, {66, 256, 128},   // sa2
        {74, 256, 261},{76, 256, 256}, {78, 512, 256},   // sa3
        {26, 256, 768},{28, 256, 256},                   // fp3
        {18, 256, 384},{20, 128, 256},                   // fp2
        {8, 128, 128}, {10, 128, 128}, {12, 128, 128},   // fp1
        {36, 100, 128},{38, 50, 100},  {40, 25, 50}, {42, 128, 25} // head
    };
    half_t* Wp[20]; int Mp16[20], Kp32[20];
    for (int i = 0; i < 20; ++i) {
        Mp16[i] = ru(wd[i].M, 16);
        Kp32[i] = ru(wd[i].K, 32);
        Wp[i] = (half_t*)ws.get((size_t)Mp16[i] * Kp32[i] * 2);
    }
    for (int i = 0; i < 20; ++i) {
        int tot = Mp16[i] * Kp32[i];
        k_pack_w<<<(tot + 255) / 256, 256, 0, stream>>>(IN(wd[i].idx), Wp[i],
                                                        wd[i].M, wd[i].K, Mp16[i], Kp32[i]);
    }

    // BN (g, beta) input indices, matching wd order (layers 0..18 have BN)
    const int bnG[19] = {44,46,48, 56,58,60, 68,70,72, 22,24, 14,16, 2,4,6, 30,32,34};
    const int bnB[19] = {43,45,47, 55,57,59, 67,69,71, 21,23, 13,15, 1,3,5, 29,31,33};

    // layer: GEMM -> stats -> BN+ReLU+transpose. rowsOut == next layer's Kp.
    auto layer = [&](int wi, const half_t* Xin, int Cols, float* Ybuf,
                     half_t* Xout, int rowsOut) {
        int M = wd[wi].M;
        dim3 g(Cols / 16, Mp16[wi] / 16);
        k_gemm<<<g, 32, 0, stream>>>(Wp[wi], Xin, Ybuf, Kp32[wi], Cols);
        k_bnstats<<<M, 256, 0, stream>>>(Ybuf, meanb, rstdb, Cols);
        dim3 gt(Cols / 32, rowsOut / 32);
        k_bnrelu_t<<<gt, 256, 0, stream>>>(Ybuf, IN(bnG[wi]), IN(bnB[wi]),
                                           meanb, rstdb, Xout, M, rowsOut, Cols);
    };

    // ---- l0 coords ----
    k_transpose_pts<<<(BB * NN + 255) / 256, 256, 0, stream>>>(x, ptsT0);

    // ---- SA1 ----
    k_fps<<<BB, 256, 0, stream>>>(ptsT0, NN, 32, fidx1, distws);
    k_gather_xyz<<<(BB * 32 * 5 + 255) / 256, 256, 0, stream>>>(l1x, ptsT0, fidx1, NN, 32);
    k_ball<<<(BB * 32 + 255) / 256, 256, 0, stream>>>(ptsT0, l1x, NN, 32, 0.04f, gidx1);
    k_group1<<<(4096 * 32 + 255) / 256, 256, 0, stream>>>(ptsT0, l1x, gidx1, Xsa1);
    layer(0, Xsa1, 4096, Ysml, Xs_a, 64);
    layer(1, Xs_a, 4096, Ysml, Xs_b, 64);
    layer(2, Xs_b, 4096, Ysml, Xs_a, 128);
    k_maxpool<<<(512 * 128 + 255) / 256, 256, 0, stream>>>(Xs_a, P1, 128, 512, 8);

    // ---- SA2 ----
    k_fps<<<BB, 256, 0, stream>>>(l1x, 32, 16, fidx2, distws);
    k_gather_xyz<<<(BB * 16 * 5 + 255) / 256, 256, 0, stream>>>(l2x, l1x, fidx2, 32, 16);
    k_ball<<<(BB * 16 + 255) / 256, 256, 0, stream>>>(l1x, l2x, 32, 16, 0.16f, gidx2);
    k_group2<<<(2048 * 160 + 255) / 256, 256, 0, stream>>>(l1x, l2x, P1, gidx2, Xsa2);
    layer(3, Xsa2, 2048, Ysml, Xs_a, 128);
    layer(4, Xs_a, 2048, Ysml, Xs_b, 128);
    layer(5, Xs_b, 2048, Ysml, Xs_a, 256);
    k_maxpool<<<(256 * 256 + 255) / 256, 256, 0, stream>>>(Xs_a, P2, 256, 256, 8);

    // ---- SA3 (group_all) ----
    k_group3<<<(256 * 288 + 255) / 256, 256, 0, stream>>>(l2x, P2, Xsa3);
    layer(6, Xsa3, 256, Ysml, Xs_a, 256);
    layer(7, Xs_a, 256, Ysml, Xs_b, 256);
    layer(8, Xs_b, 256, Ysml, Xs_a, 512);
    k_maxpool<<<(16 * 512 + 255) / 256, 256, 0, stream>>>(Xs_a, F3, 512, 16, 16);

    // ---- FP3 (S==1 broadcast) ----
    k_copy_rows<<<(256 * 256 + 255) / 256, 256, 0, stream>>>(P2, Xfp3, 256, 256, 768, 0, 256);
    k_bcast_f3<<<(256 * 512 + 255) / 256, 256, 0, stream>>>(F3, Xfp3);
    layer(9, Xfp3, 256, Ysml, Xs_a, 256);
    layer(10, Xs_a, 256, Ysml, L2F, 256);

    // ---- FP2 (3-NN interp from l2 to l1) ----
    {
        dim3 g(1, BB);
        k_knn3<<<g, 256, 0, stream>>>(l1x, l2x, 32, 16, idx3s, w3s);
    }
    k_copy_rows<<<(128 * 512 + 255) / 256, 256, 0, stream>>>(P1, Xfp2, 128, 128, 384, 0, 512);
    {
        long tot = (long)256 * BB * 32;
        k_interp_build<<<(unsigned)((tot + 255) / 256), 256, 0, stream>>>(
            L2F, 256, idx3s, w3s, Xfp2, 384, 128, 256, 32, 16);
    }
    layer(11, Xfp2, 512, Ysml, Xs_a, 256);
    layer(12, Xs_a, 512, Ysml, L1F, 128);

    // ---- FP1 (3-NN interp from l1 to l0, big) ----
    {
        dim3 g(NN / 256, BB);
        k_knn3<<<g, 256, 0, stream>>>(ptsT0, l1x, NN, 32, idx3b, w3b);
    }
    {
        long tot = (long)128 * BB * NN;
        k_interp_build<<<(unsigned)((tot + 255) / 256), 256, 0, stream>>>(
            L1F, 128, idx3b, w3b, Xbig_a, 128, 0, 128, NN, 32);
    }
    const int ColsB = BB * NN;
    layer(13, Xbig_a, ColsB, Ybig, Xbig_b, 128);
    layer(14, Xbig_b, ColsB, Ybig, Xbig_a, 128);
    layer(15, Xbig_a, ColsB, Ybig, Xbig_b, 128);

    // ---- head ----
    layer(16, Xbig_b, ColsB, Ybig, Xbig_a, 128);  // 128->100 (rows padded to 128)
    layer(17, Xbig_a, ColsB, Ybig, Xbig_b, 64);   // 100->50
    layer(18, Xbig_b, ColsB, Ybig, Xbig_a, 32);   // 50->25
    {
        dim3 g(ColsB / 16, Mp16[19] / 16);        // 25->128, logits
        k_gemm<<<g, 32, 0, stream>>>(Wp[19], Xbig_a, Ybig, Kp32[19], ColsB);
    }
    k_logsoftmax<<<(BB * NN + 255) / 256, 256, 0, stream>>>(Ybig, IN(41), (float*)d_out);
}
